// ProteinMPNN_47777216201140
// MI455X (gfx1250) — matrix operand order repaired and checked by reading, not compiled, not run on hardware
//
#include <hip/hip_runtime.h>
#include <hip/hip_bf16.h>

typedef __attribute__((ext_vector_type(16))) _Float16 v16h;
typedef __attribute__((ext_vector_type(8)))  _Float16 v8h;
typedef __attribute__((ext_vector_type(8)))  float    v8f;
typedef __attribute__((ext_vector_type(4)))  float    v4f;

#define HB   2
#define NN   2048
#define KN   48
#define HD   128
#define FFD  512
#define EPSL 1e-5f
#define RSCALE (1.0f/30.0f)

// Exact-erf GELU via branch-free Abramowitz-Stegun 7.1.26 rational approx
// (max abs err ~1.5e-7, well below the f16-GEMM noise floor). No EXEC
// divergence: one v_rcp_f32 + one v_exp_f32 + straight-line FMAs.
__device__ __forceinline__ float gelu_ex(float x){
  float ax = fabsf(x) * 0.70710678118654752440f;     // |x|/sqrt(2)
  float t  = __builtin_amdgcn_rcpf(fmaf(0.3275911f, ax, 1.0f));
  float p  = t*(0.254829592f + t*(-0.284496736f + t*(1.421413741f +
             t*(-1.453152027f + t*1.061405429f))));
  float er = 1.0f - p * __expf(-ax*ax);              // erf(|x|/sqrt2)
  float ev = copysignf(er, x);
  return 0.5f * x * (1.0f + ev);
}
__device__ __forceinline__ v8f splat8(float x){
  v8f r;
#pragma unroll
  for(int i=0;i<8;i++) r[i]=x;
  return r;
}
__device__ __forceinline__ v8f wmma16(v16h a, v16h b, v8f c){
  return __builtin_amdgcn_wmma_f32_16x16x32_f16(false, a, false, b, (short)0, c, false, false);
}
// A-fragment (16x32 f16) from a row of f32 features.
// Lanes 0-15 hold K = {0..7,16..23}, lanes 16-31 hold K = {8..15,24..31}; c0 = base + hh*8.
__device__ __forceinline__ v16h afrag_g(const float* __restrict__ row, int c0){
  v4f x0 = *(const v4f*)(row + c0);
  v4f x1 = *(const v4f*)(row + c0 + 4);
  v4f y0 = *(const v4f*)(row + c0 + 16);
  v4f y1 = *(const v4f*)(row + c0 + 20);
  v16h a;
#pragma unroll
  for(int i=0;i<4;i++){
    a[i]      = (_Float16)x0[i];
    a[4+i]    = (_Float16)x1[i];
    a[8+i]    = (_Float16)y0[i];
    a[12+i]   = (_Float16)y1[i];
  }
  return a;
}
// A-fragment from LDS staging; abase = stg + m*SLD + hh*8 (row-internal kc offsets)
__device__ __forceinline__ v16h afrag_sp(const _Float16* abase, int kc){
  v8h lo = *(const v8h*)(abase + kc*32);
  v8h hi = *(const v8h*)(abase + kc*32 + 16);
  v16h a;
#pragma unroll
  for(int i=0;i<8;i++){ a[i]=lo[i]; a[8+i]=hi[i]; }
  return a;
}
// B-fragment: wcol = Wt + col*kd + hh*8 ; halves at +0..7 and +16..23 per lane
__device__ __forceinline__ v16h bfrag_p(const _Float16* __restrict__ wcol, int kc){
  v8h lo = *(const v8h*)(wcol + kc*32);
  v8h hi = *(const v8h*)(wcol + kc*32 + 16);
  v16h b;
#pragma unroll
  for(int i=0;i<8;i++){ b[i]=lo[i]; b[8+i]=hi[i]; }
  return b;
}

// ---------------- K0: weight transpose + f32->f16 convert -------------------
__global__ __launch_bounds__(256) void k_wcvt(const float* __restrict__ src,
                                              _Float16* __restrict__ dst,
                                              int Kd, int Nd){
  int i = blockIdx.x*256 + threadIdx.x;
  if(i >= Kd*Nd) return;
  int n = i / Kd, k = i - n*Kd;
  dst[i] = (_Float16)src[(size_t)k*Nd + n];
}

// ---------------- K1: node message MLP + masked K-sum -> dh -----------------
__global__ __launch_bounds__(256) void k_node_msg(
    const float* __restrict__ hV, const float* __restrict__ hE,
    const int*   __restrict__ Eidx, const float* __restrict__ mAtt,
    const _Float16* __restrict__ W1t, const _Float16* __restrict__ W2t,
    const _Float16* __restrict__ W3t,
    const float* __restrict__ b1, const float* __restrict__ b2,
    const float* __restrict__ b3,
    float* __restrict__ dh)
{
  extern __shared__ _Float16 smem[];
  const int SLD = 136;
  const int wave = threadIdx.x >> 5;
  const int lane = threadIdx.x & 31;
  _Float16* stgA = smem + wave * (2*16*SLD);
  _Float16* stgB = stgA + 16*SLD;
  const int node = blockIdx.x * 8 + wave;      // flat b*N+n
  const int bb   = node / NN;
  const int m    = lane & 15;
  const int hh   = lane >> 4;
  const float* rowV = hV + (size_t)node * HD;
  const _Float16* aA = stgA + m*SLD + hh*8;
  const _Float16* aB = stgB + m*SLD + hh*8;
  float dhacc[8];
#pragma unroll
  for(int i=0;i<8;i++) dhacc[i]=0.f;

  for(int t=0;t<3;t++){
    const int ke = t*16 + m;
    const float* rowE = hE + ((size_t)node*KN + ke)*HD;
    const int gi = Eidx[(size_t)node*KN + ke];
    const float* rowG = hV + ((size_t)bb*NN + gi)*HD;
    // ---- layer 1: [16x384]x[384x128] -> gelu -> stgA (two groups of 4 cols)
#pragma unroll
    for(int g=0;g<2;g++){
      v8f acc[4]; const _Float16* wp[4];
#pragma unroll
      for(int nt=0;nt<4;nt++){
        int col=(g*4+nt)*16+m;
        acc[nt]=splat8(b1[col]);
        wp[nt]=W1t + (size_t)col*384 + hh*8;
      }
#pragma unroll 1
      for(int kc=0;kc<12;kc++){
        const float* src = (kc<4)? rowV : (kc<8 ? rowE : rowG);
        v16h a = afrag_g(src, (kc&3)*32 + hh*8);
#pragma unroll
        for(int nt=0;nt<4;nt++)
          acc[nt] = wmma16(a, bfrag_p(wp[nt],kc), acc[nt]);
      }
#pragma unroll
      for(int nt=0;nt<4;nt++)
#pragma unroll
        for(int j=0;j<8;j++)
          stgA[(hh*8+j)*SLD + (g*4+nt)*16+m] = (_Float16)gelu_ex(acc[nt][j]);
    }
    __builtin_amdgcn_wave_barrier();
    // ---- layer 2: stgA -> gelu -> stgB
#pragma unroll
    for(int g=0;g<2;g++){
      v8f acc[4]; const _Float16* wp[4];
#pragma unroll
      for(int nt=0;nt<4;nt++){
        int col=(g*4+nt)*16+m;
        acc[nt]=splat8(b2[col]);
        wp[nt]=W2t + (size_t)col*128 + hh*8;
      }
#pragma unroll 1
      for(int kc=0;kc<4;kc++){
        v16h a = afrag_sp(aA,kc);
#pragma unroll
        for(int nt=0;nt<4;nt++)
          acc[nt] = wmma16(a, bfrag_p(wp[nt],kc), acc[nt]);
      }
#pragma unroll
      for(int nt=0;nt<4;nt++)
#pragma unroll
        for(int j=0;j<8;j++)
          stgB[(hh*8+j)*SLD + (g*4+nt)*16+m] = (_Float16)gelu_ex(acc[nt][j]);
    }
    __builtin_amdgcn_wave_barrier();
    // ---- layer 3: stgB -> masked K-partial-sum into dhacc
    float msk[8];
#pragma unroll
    for(int j=0;j<8;j++) msk[j] = mAtt[(size_t)node*KN + t*16 + hh*8 + j];
#pragma unroll
    for(int g=0;g<2;g++){
      v8f acc[4]; const _Float16* wp[4];
#pragma unroll
      for(int nt=0;nt<4;nt++){
        int col=(g*4+nt)*16+m;
        acc[nt]=splat8(b3[col]);
        wp[nt]=W3t + (size_t)col*128 + hh*8;
      }
#pragma unroll 1
      for(int kc=0;kc<4;kc++){
        v16h a = afrag_sp(aB,kc);
#pragma unroll
        for(int nt=0;nt<4;nt++)
          acc[nt] = wmma16(a, bfrag_p(wp[nt],kc), acc[nt]);
      }
#pragma unroll
      for(int nt=0;nt<4;nt++){
        float s = 0.f;
#pragma unroll
        for(int j=0;j<8;j++) s += msk[j]*acc[nt][j];
        dhacc[g*4+nt] += s;
      }
    }
    __builtin_amdgcn_wave_barrier();
  }
#pragma unroll
  for(int nt=0;nt<8;nt++){
    float s = dhacc[nt] + __shfl_xor(dhacc[nt], 16, 32);
    if(lane < 16) dh[(size_t)node*HD + nt*16 + lane] = s;
  }
}

// ---------------- K2: LN1 -> FFN (WMMA) -> LN2 -> mask_V --------------------
__global__ __launch_bounds__(256) void k_node_upd(
    const float* __restrict__ hV, const float* __restrict__ dh,
    const float* __restrict__ mV,
    const _Float16* __restrict__ Wit, const _Float16* __restrict__ Wot,
    const float* __restrict__ bi, const float* __restrict__ bo,
    const float* __restrict__ g1, const float* __restrict__ be1,
    const float* __restrict__ g2, const float* __restrict__ be2,
    float* __restrict__ hVout)
{
  extern __shared__ _Float16 smem[];
  const int S1 = 136, S2 = 520;
  const int wave = threadIdx.x >> 5;
  const int lane = threadIdx.x & 31;
  _Float16* s1 = smem + wave * (16*S1 + 16*S2);
  _Float16* s2 = s1 + 16*S1;
  const int nb = (blockIdx.x*8 + wave) * 16;
  const int m = lane & 15, hh = lane >> 4;
  // ---- LN1 (two-pass): lane handles node row m, cols [hh*64, hh*64+64)
  {
    const int node = nb + m;
    const float* xr = hV + (size_t)node*HD;
    const float* dr = dh + (size_t)node*HD;
    const int c0 = hh*64;
    float s=0.f, q=0.f;
#pragma unroll
    for(int i=0;i<16;i++){
      v4f a = *(const v4f*)(xr + c0 + i*4);
      v4f d = *(const v4f*)(dr + c0 + i*4);
#pragma unroll
      for(int k2=0;k2<4;k2++){
        float t = a[k2] + d[k2]*RSCALE;
        s+=t; q+=t*t;
      }
    }
    s += __shfl_xor(s,16,32); q += __shfl_xor(q,16,32);
    float mean = s*(1.f/HD);
    float rstd = rsqrtf(q*(1.f/HD) - mean*mean + EPSL);
#pragma unroll
    for(int i=0;i<8;i++){
      v4f a0 = *(const v4f*)(xr + c0 + i*8);
      v4f d0 = *(const v4f*)(dr + c0 + i*8);
      v4f a1 = *(const v4f*)(xr + c0 + i*8 + 4);
      v4f d1 = *(const v4f*)(dr + c0 + i*8 + 4);
      v8h h;
#pragma unroll
      for(int k2=0;k2<4;k2++){
        int c = c0 + i*8 + k2;
        h[k2]   = (_Float16)(((a0[k2]+d0[k2]*RSCALE)-mean)*rstd*g1[c]   + be1[c]);
        h[4+k2] = (_Float16)(((a1[k2]+d1[k2]*RSCALE)-mean)*rstd*g1[c+4] + be1[c+4]);
      }
      *(v8h*)(s1 + m*S1 + c0 + i*8) = h;
    }
  }
  __builtin_amdgcn_wave_barrier();
  // ---- FFN in: [16x128]x[128x512], GELU, stage in s2
  const _Float16* aA = s1 + m*S1 + hh*8;
#pragma unroll 1
  for(int nt=0;nt<32;nt++){
    const _Float16* wcol = Wit + (size_t)(nt*16+m)*128 + hh*8;
    v8f acc = splat8(bi[nt*16+m]);
#pragma unroll
    for(int kc=0;kc<4;kc++)
      acc = wmma16(afrag_sp(aA,kc), bfrag_p(wcol,kc), acc);
#pragma unroll
    for(int j=0;j<8;j++)
      s2[(hh*8+j)*S2 + nt*16+m] = (_Float16)gelu_ex(acc[j]);
  }
  __builtin_amdgcn_wave_barrier();
  // ---- FFN out: [16x512]x[512x128] (two groups) + residual stash
  const _Float16* aB = s2 + m*S2 + hh*8;
  float x[8][8], rs[8], rq[8];
#pragma unroll
  for(int j=0;j<8;j++){ rs[j]=0.f; rq[j]=0.f; }
#pragma unroll
  for(int g=0;g<2;g++){
    v8f acc[4]; const _Float16* wp[4];
#pragma unroll
    for(int nt=0;nt<4;nt++){
      int col=(g*4+nt)*16+m;
      acc[nt]=splat8(bo[col]);
      wp[nt]=Wot + (size_t)col*512 + hh*8;
    }
#pragma unroll 1
    for(int kc=0;kc<16;kc++){
      v16h a = afrag_sp(aB,kc);
#pragma unroll
      for(int nt=0;nt<4;nt++)
        acc[nt] = wmma16(a, bfrag_p(wp[nt],kc), acc[nt]);
    }
#pragma unroll
    for(int nt=0;nt<4;nt++)
#pragma unroll
      for(int j=0;j<8;j++){
        float h1v = (float)s1[(j+8*hh)*S1 + (g*4+nt)*16+m];
        float v = h1v + acc[nt][j];
        x[g*4+nt][j]=v; rs[j]+=v; rq[j]+=v*v;
      }
  }
#pragma unroll
  for(int j=0;j<8;j++){
#pragma unroll
    for(int d2=1; d2<16; d2<<=1){
      rs[j]+=__shfl_xor(rs[j],d2,32);
      rq[j]+=__shfl_xor(rq[j],d2,32);
    }
  }
#pragma unroll
  for(int j=0;j<8;j++){
    float mean = rs[j]*(1.f/HD);
    float rstd = rsqrtf(rq[j]*(1.f/HD)-mean*mean+EPSL);
    int node = nb + j + 8*hh;
    float mk = mV[node];
    float* orow = hVout + (size_t)node*HD;
#pragma unroll
    for(int nt=0;nt<8;nt++){
      int c = nt*16+m;
      orow[c] = ((x[nt][j]-mean)*rstd*g2[c]+be2[c])*mk;
    }
  }
}

// ---------------- K3: edge message MLP + residual LN3 -> h_E out ------------
__global__ __launch_bounds__(256) void k_edge_upd(
    const float* __restrict__ hVn, const float* __restrict__ hE,
    const int* __restrict__ Eidx,
    const _Float16* __restrict__ W11t, const _Float16* __restrict__ W12t,
    const _Float16* __restrict__ W13t,
    const float* __restrict__ b11, const float* __restrict__ b12,
    const float* __restrict__ b13,
    const float* __restrict__ g3, const float* __restrict__ be3,
    float* __restrict__ outE)
{
  extern __shared__ _Float16 smem[];
  const int SLD=136;
  const int wave=threadIdx.x>>5, lane=threadIdx.x&31;
  _Float16* stgA = smem + wave*(2*16*SLD);
  _Float16* stgB = stgA + 16*SLD;
  const int node = blockIdx.x*8 + wave;
  const int bb = node / NN;
  const int m=lane&15, hh=lane>>4;
  const float* rowV = hVn + (size_t)node*HD;
  const _Float16* aA = stgA + m*SLD + hh*8;
  const _Float16* aB = stgB + m*SLD + hh*8;
  for(int t=0;t<3;t++){
    const int ke = t*16+m;
    const float* rowE = hE + ((size_t)node*KN+ke)*HD;
    const int gi = Eidx[(size_t)node*KN+ke];
    const float* rowG = hVn + ((size_t)bb*NN+gi)*HD;
    // ---- layer 1 -> stgA
#pragma unroll
    for(int g=0;g<2;g++){
      v8f acc[4]; const _Float16* wp[4];
#pragma unroll
      for(int nt=0;nt<4;nt++){
        int col=(g*4+nt)*16+m;
        acc[nt]=splat8(b11[col]);
        wp[nt]=W11t + (size_t)col*384 + hh*8;
      }
#pragma unroll 1
      for(int kc=0;kc<12;kc++){
        const float* src = (kc<4)? rowV : (kc<8 ? rowE : rowG);
        v16h a = afrag_g(src, (kc&3)*32 + hh*8);
#pragma unroll
        for(int nt=0;nt<4;nt++)
          acc[nt] = wmma16(a, bfrag_p(wp[nt],kc), acc[nt]);
      }
#pragma unroll
      for(int nt=0;nt<4;nt++)
#pragma unroll
        for(int j=0;j<8;j++)
          stgA[(hh*8+j)*SLD + (g*4+nt)*16+m] = (_Float16)gelu_ex(acc[nt][j]);
    }
    __builtin_amdgcn_wave_barrier();
    // ---- layer 2: stgA -> stgB
#pragma unroll
    for(int g=0;g<2;g++){
      v8f acc[4]; const _Float16* wp[4];
#pragma unroll
      for(int nt=0;nt<4;nt++){
        int col=(g*4+nt)*16+m;
        acc[nt]=splat8(b12[col]);
        wp[nt]=W12t + (size_t)col*128 + hh*8;
      }
#pragma unroll 1
      for(int kc=0;kc<4;kc++){
        v16h a = afrag_sp(aA,kc);
#pragma unroll
        for(int nt=0;nt<4;nt++)
          acc[nt] = wmma16(a, bfrag_p(wp[nt],kc), acc[nt]);
      }
#pragma unroll
      for(int nt=0;nt<4;nt++)
#pragma unroll
        for(int j=0;j<8;j++)
          stgB[(hh*8+j)*SLD + (g*4+nt)*16+m] = (_Float16)gelu_ex(acc[nt][j]);
    }
    __builtin_amdgcn_wave_barrier();
    // ---- layer 3: stgB -> residual stash
    const int kb = t*16 + hh*8;
    const float* erb = hE   + ((size_t)node*KN + kb)*HD;
    float x[8][8], rs[8], rq[8];
#pragma unroll
    for(int j=0;j<8;j++){ rs[j]=0.f; rq[j]=0.f; }
#pragma unroll
    for(int g=0;g<2;g++){
      v8f acc[4]; const _Float16* wp[4];
#pragma unroll
      for(int nt=0;nt<4;nt++){
        int col=(g*4+nt)*16+m;
        acc[nt]=splat8(b13[col]);
        wp[nt]=W13t + (size_t)col*128 + hh*8;
      }
#pragma unroll 1
      for(int kc=0;kc<4;kc++){
        v16h a = afrag_sp(aB,kc);
#pragma unroll
        for(int nt=0;nt<4;nt++)
          acc[nt] = wmma16(a, bfrag_p(wp[nt],kc), acc[nt]);
      }
#pragma unroll
      for(int nt=0;nt<4;nt++)
#pragma unroll
        for(int j=0;j<8;j++){
          int c=(g*4+nt)*16+m;
          float v = erb[j*HD + c] + acc[nt][j];
          x[g*4+nt][j]=v; rs[j]+=v; rq[j]+=v*v;
        }
    }
#pragma unroll
    for(int j=0;j<8;j++){
#pragma unroll
      for(int d2=1; d2<16; d2<<=1){
        rs[j]+=__shfl_xor(rs[j],d2,32);
        rq[j]+=__shfl_xor(rq[j],d2,32);
      }
    }
    float* ob = outE + ((size_t)node*KN + kb)*HD;
#pragma unroll
    for(int j=0;j<8;j++){
      float mean = rs[j]*(1.f/HD);
      float rstd = rsqrtf(rq[j]*(1.f/HD)-mean*mean+EPSL);
#pragma unroll
      for(int nt=0;nt<8;nt++){
        int c = nt*16+m;
        ob[j*HD + c] = (x[nt][j]-mean)*rstd*g3[c]+be3[c];
      }
    }
    __builtin_amdgcn_wave_barrier();
  }
}

extern "C" void kernel_launch(void* const* d_in, const int* in_sizes, int n_in,
                              void* d_out, int out_size, void* d_ws, size_t ws_size,
                              hipStream_t stream)
{
  const float* hV  = (const float*)d_in[0];
  const float* hE  = (const float*)d_in[1];
  const int*   Eix = (const int*)d_in[2];
  const float* mV  = (const float*)d_in[3];
  const float* mA  = (const float*)d_in[4];
  const float* W1  = (const float*)d_in[5];  const float* b1 = (const float*)d_in[6];
  const float* W2  = (const float*)d_in[7];  const float* b2 = (const float*)d_in[8];
  const float* W3  = (const float*)d_in[9];  const float* b3 = (const float*)d_in[10];
  const float* W11 = (const float*)d_in[11]; const float* b11= (const float*)d_in[12];
  const float* W12 = (const float*)d_in[13]; const float* b12= (const float*)d_in[14];
  const float* W13 = (const float*)d_in[15]; const float* b13= (const float*)d_in[16];
  const float* Wi  = (const float*)d_in[17]; const float* bi = (const float*)d_in[18];
  const float* Wo  = (const float*)d_in[19]; const float* bo = (const float*)d_in[20];
  const float* g1  = (const float*)d_in[21]; const float* be1= (const float*)d_in[22];
  const float* g2  = (const float*)d_in[23]; const float* be2= (const float*)d_in[24];
  const float* g3  = (const float*)d_in[25]; const float* be3= (const float*)d_in[26];

  // workspace layout: f16 transposed weights [0, 589824B), dh f32 at +1MB (2MB)
  _Float16* wf = (_Float16*)d_ws;
  _Float16* W1t  = wf + 0;
  _Float16* W2t  = wf + 49152;
  _Float16* W3t  = wf + 65536;
  _Float16* W11t = wf + 81920;
  _Float16* W12t = wf + 131072;
  _Float16* W13t = wf + 147456;
  _Float16* Wit  = wf + 163840;
  _Float16* Wot  = wf + 229376;
  float* dh = (float*)((char*)d_ws + (1u<<20));

  float* hVout = (float*)d_out;                        // [B,N,H]
  float* outE  = hVout + (size_t)HB*NN*HD;             // [B,N,K,H]

  k_wcvt<<<192,256,0,stream>>>(W1, W1t, 384,128);
  k_wcvt<<< 64,256,0,stream>>>(W2, W2t, 128,128);
  k_wcvt<<< 64,256,0,stream>>>(W3, W3t, 128,128);
  k_wcvt<<<192,256,0,stream>>>(W11,W11t,384,128);
  k_wcvt<<< 64,256,0,stream>>>(W12,W12t,128,128);
  k_wcvt<<< 64,256,0,stream>>>(W13,W13t,128,128);
  k_wcvt<<<256,256,0,stream>>>(Wi, Wit, 128,512);
  k_wcvt<<<256,256,0,stream>>>(Wo, Wot, 512,128);

  const int shm1 = 8*(2*16*136)*2;        // 69,632 B (ping-pong stage)
  const int shm2 = 8*16*(136+520)*2;      // 167,936 B
  k_node_msg<<<HB*NN/8, 256, shm1, stream>>>(hV,hE,Eix,mA,W1t,W2t,W3t,b1,b2,b3,dh);
  k_node_upd<<<HB*NN/128, 256, shm2, stream>>>(hV,dh,mV,Wit,Wot,bi,bo,g1,be1,g2,be2,hVout);
  k_edge_upd<<<HB*NN/8, 256, shm1, stream>>>(hVout,hE,Eix,W11t,W12t,W13t,b11,b12,b13,g3,be3,outE);
}
